// Model_39676907881550
// MI455X (gfx1250) — matrix operand order repaired and checked by reading, not compiled, not run on hardware
//
#include <hip/hip_runtime.h>

// ---------------- problem constants ----------------
#define BATCH 16
#define S_LEN 2048
#define D_DIM 256
#define QT 32              // q rows per attention block
#define KT 32              // keys per attention iteration
#define SCALE 0.0625f      // 256^-0.5

typedef __attribute__((ext_vector_type(16))) __bf16 v16bf;
typedef __attribute__((ext_vector_type(8)))  float  v8f;
typedef __attribute__((ext_vector_type(4)))  int    v4i;

#define AS1 __attribute__((address_space(1)))
#define AS3 __attribute__((address_space(3)))

// gfx1250 async memory->LDS path (ASYNCcnt), if the toolchain exposes it.
#if __has_builtin(__builtin_amdgcn_global_load_async_to_lds_b128) && \
    __has_builtin(__builtin_amdgcn_s_wait_asynccnt)
#define HAVE_ASYNC_LDS 1
#else
#define HAVE_ASYNC_LDS 0
#endif

// Contiguous tile copy global->LDS, 128 threads, n16B uint4 elements.
__device__ __forceinline__ void stage_tile(const __bf16* __restrict__ g,
                                           __bf16* l, int n16B, int tid) {
  const uint4* src = (const uint4*)g;
  uint4* dst = (uint4*)l;
#if HAVE_ASYNC_LDS
  for (int i = tid; i < n16B; i += 128)
    __builtin_amdgcn_global_load_async_to_lds_b128(
        (AS1 v4i*)(uint4*)(src + i), (AS3 v4i*)(dst + i), 0, 0);
#else
  for (int i = tid; i < n16B; i += 128) dst[i] = src[i];
#endif
}

__device__ __forceinline__ void stage_wait() {
#if HAVE_ASYNC_LDS
  __builtin_amdgcn_s_wait_asynccnt(0);
#endif
}

// float -> bf16, round-to-nearest-even
__device__ __forceinline__ __bf16 f2bf(float f) {
  union { float f; unsigned u; } a; a.f = f;
  unsigned r = (a.u + 0x7FFFu + ((a.u >> 16) & 1u)) >> 16;
  union { unsigned short s; __bf16 b; } o; o.s = (unsigned short)r;
  return o.b;
}

// Load 16 bf16 as two contiguous 8-element runs (each 16B aligned).
__device__ __forceinline__ v16bf load16bf(const __bf16* p0, const __bf16* p1) {
  union { uint4 q[2]; v16bf v; } u;
  u.q[0] = *(const uint4*)p0;
  u.q[1] = *(const uint4*)p1;
  return u.v;
}

// ---------------- kernel 0: convert weights to bf16 ----------------
__global__ void cvt_w_kernel(const float* __restrict__ wq,
                             const float* __restrict__ wk,
                             const float* __restrict__ wv,
                             __bf16* __restrict__ oq,
                             __bf16* __restrict__ ok,
                             __bf16* __restrict__ ov) {
  int i = blockIdx.x * blockDim.x + threadIdx.x;
  const int N = D_DIM * D_DIM;  // 65536
  if (i < N)            oq[i]         = f2bf(wq[i]);
  else if (i < 2 * N)   ok[i - N]     = f2bf(wk[i - N]);
  else if (i < 3 * N)   ov[i - 2 * N] = f2bf(wv[i - 2 * N]);
}

// ---------------- kernel 1: fused QKV projection ----------------
// Per output 16x16 tile: preload all 8 B-chunks (global_load_b128 pairs,
// clause-friendly), then 8 back-to-back WMMAs.
__device__ __forceinline__ void proj_one_matrix(
    const v16bf* xa,                    // A-layout x fragments (8 chunks)
    const __bf16* __restrict__ W,       // bf16 weights, row-major [e][d]
    const float*  __restrict__ bias,
    __bf16*       __restrict__ O,       // bf16 output [b*s][d]
    int wave, int l16, int half, long row0) {
  for (int nt = wave; nt < 16; nt += 4) {
    // B layout: lane holds K=j+half*16 (j=0..15) of column n = nt*16+l16
    // B[k][n] = W[n][k] -> 16 contiguous bf16 per chunk
    const __bf16* wr = W + (size_t)(nt * 16 + l16) * D_DIM + half * 16;
    v16bf bm[8];
#pragma unroll
    for (int c = 0; c < 8; ++c)
      bm[c] = load16bf(wr + c * 32, wr + c * 32 + 8);
    v8f acc = {};
#pragma unroll
    for (int c = 0; c < 8; ++c)
      acc = __builtin_amdgcn_wmma_f32_16x16x32_bf16(
          false, xa[c], false, bm[c], (short)0, acc, false, false);
    const float bv = bias[nt * 16 + l16];
#pragma unroll
    for (int r = 0; r < 8; ++r) {
      const long row = row0 + r + half * 8;   // C layout: M = r + half*8
      O[row * D_DIM + nt * 16 + l16] = f2bf(acc[r] + bv);
    }
  }
}

__global__ __launch_bounds__(128)
void proj_kernel(const float* __restrict__ x,
                 const __bf16* __restrict__ wq, const float* __restrict__ bq,
                 const __bf16* __restrict__ wk, const float* __restrict__ bk,
                 const __bf16* __restrict__ wv, const float* __restrict__ bv,
                 __bf16* __restrict__ qo, __bf16* __restrict__ ko,
                 __bf16* __restrict__ vo) {
  __shared__ __bf16 lds_x[16 * D_DIM];   // 8 KB

  const int tid  = threadIdx.x;
  const int lane = tid & 31;
  const int wave = tid >> 5;
  const int half = lane >> 4;   // which 16-lane half
  const int l16  = lane & 15;
  const long row0 = (long)blockIdx.x * 16;

  // stage x tile as bf16 into LDS (row-major 16 x 256)
  {
    const float4* src = (const float4*)(x + row0 * D_DIM);
    for (int i = tid; i < 16 * D_DIM / 4; i += 128) {
      float4 f = src[i];
      __bf16* d = lds_x + i * 4;
      d[0] = f2bf(f.x); d[1] = f2bf(f.y); d[2] = f2bf(f.z); d[3] = f2bf(f.w);
    }
  }
  __syncthreads();

  // A-layout x: 8 chunks of 16x32; per-lane two contiguous 8-elem runs
  v16bf xa[8];
  {
    const __bf16* xr = lds_x + l16 * D_DIM;
#pragma unroll
    for (int c = 0; c < 8; ++c)
      xa[c] = load16bf(xr + c * 32 + half * 16, xr + c * 32 + half * 16 + 8);
  }

  // static per-matrix calls: keeps global address space + restrict info
  proj_one_matrix(xa, wq, bq, qo, wave, l16, half, row0);
  proj_one_matrix(xa, wk, bk, ko, wave, l16, half, row0);
  proj_one_matrix(xa, wv, bv, vo, wave, l16, half, row0);
}

// ---------------- kernel 2: flash attention ----------------
// Block: 128 threads = 4 waves = (2 q-subtiles) x (2 column halves).
// 32 q-rows per block; streams 32-key K/V tiles through LDS with online
// softmax in fp32; out written fp32.
__global__ __launch_bounds__(128)
void attn_kernel(const __bf16* __restrict__ qg,
                 const __bf16* __restrict__ kg,
                 const __bf16* __restrict__ vg,
                 float* __restrict__ out) {
  __shared__ __bf16 lds_q[QT * D_DIM];      // 16 KB
  __shared__ __bf16 lds_k[KT * D_DIM];      // 16 KB
  __shared__ __bf16 lds_vT[D_DIM * KT];     // 16 KB (transposed: [d][key])
  __shared__ __bf16 lds_p[4 * 16 * KT];     // 4 KB  (per-wave P scratch)

  const int tid  = threadIdx.x;
  const int lane = tid & 31;
  const int wave = tid >> 5;
  const int wm   = wave >> 1;   // q sub-tile (0..1)
  const int wn   = wave & 1;    // output column half (0..1)
  const int half = lane >> 4;
  const int l16  = lane & 15;

  const int  b     = blockIdx.x;
  const int  qt    = blockIdx.y;
  const long qrow0 = (long)b * S_LEN + (long)qt * QT;

  // stage Q tile (contiguous 16 KB) -- async DMA to LDS when available
  stage_tile(qg + qrow0 * D_DIM, lds_q, QT * D_DIM / 8, tid);
  stage_wait();
  __syncthreads();

  // Q A-fragments are invariant across all key iterations: keep in VGPRs
  v16bf qa[8];
  {
    const __bf16* qrow = lds_q + (wm * 16 + l16) * D_DIM;
#pragma unroll
    for (int c = 0; c < 8; ++c)
      qa[c] = load16bf(qrow + c * 32 + half * 16,
                       qrow + c * 32 + half * 16 + 8);
  }

  v8f o[8];
#pragma unroll
  for (int t = 0; t < 8; ++t) o[t] = (v8f){};
  float m_s[8], l_s[8];
#pragma unroll
  for (int r = 0; r < 8; ++r) { m_s[r] = -1e30f; l_s[r] = 0.0f; }

  __bf16* pw = lds_p + wave * 16 * KT;

  const int nIter = S_LEN / KT;   // 64
  for (int it = 0; it < nIter; ++it) {
    const long krow0 = (long)b * S_LEN + (long)it * KT;
    __syncthreads();   // previous iteration's LDS reads done
    // stage K tile (async contiguous DMA) and transposed V tile
    {
      stage_tile(kg + krow0 * D_DIM, lds_k, KT * D_DIM / 8, tid);
      const uint4* vs = (const uint4*)(vg + krow0 * D_DIM);
      for (int i = tid; i < KT * D_DIM / 8; i += 128) {
        union { uint4 u; __bf16 h[8]; } cv; cv.u = vs[i];
        const int krow = i >> 5;            // 32 uint4 per 256-wide row
        const int d0   = (i & 31) * 8;
#pragma unroll
        for (int e = 0; e < 8; ++e) lds_vT[(d0 + e) * KT + krow] = cv.h[e];
      }
      if (it + 1 < nIter) {   // gfx1250 global_prefetch_b8 for next tiles
        __builtin_prefetch(kg + (krow0 + KT) * D_DIM, 0, 1);
        __builtin_prefetch(vg + (krow0 + KT) * D_DIM, 0, 1);
      }
    }
    stage_wait();
    __syncthreads();

    // ---- scores: S = Q Kt, two 16x16 key tiles, K-dim 256 in 8 WMMA steps
    v8f s0 = (v8f){}, s1 = (v8f){};
    {
      // B[k][n] = K[key n][d k]: 16 contiguous bf16 per lane per chunk
      const __bf16* kb0 = lds_k + (0 * 16 + l16) * D_DIM + half * 16;
      const __bf16* kb1 = lds_k + (1 * 16 + l16) * D_DIM + half * 16;
#pragma unroll
      for (int c = 0; c < 8; ++c) {
        v16bf b0 = load16bf(kb0 + c * 32, kb0 + c * 32 + 8);
        s0 = __builtin_amdgcn_wmma_f32_16x16x32_bf16(
            false, qa[c], false, b0, (short)0, s0, false, false);
        v16bf b1 = load16bf(kb1 + c * 32, kb1 + c * 32 + 8);
        s1 = __builtin_amdgcn_wmma_f32_16x16x32_bf16(
            false, qa[c], false, b1, (short)0, s1, false, false);
      }
    }

    // ---- online softmax (fp32). Row r+half*8 lives across 16 lanes.
    float alpha[8];
#pragma unroll
    for (int r = 0; r < 8; ++r) {
      float a0 = s0[r] * SCALE;
      float a1 = s1[r] * SCALE;
      float mx = fmaxf(a0, a1);
#pragma unroll
      for (int msk = 8; msk; msk >>= 1)
        mx = fmaxf(mx, __shfl_xor(mx, msk, 16));
      const float mnew = fmaxf(m_s[r], mx);
      const float al   = __expf(m_s[r] - mnew);
      const float p0   = __expf(a0 - mnew);
      const float p1   = __expf(a1 - mnew);
      float rs = p0 + p1;
#pragma unroll
      for (int msk = 8; msk; msk >>= 1)
        rs += __shfl_xor(rs, msk, 16);
      l_s[r]  = l_s[r] * al + rs;
      m_s[r]  = mnew;
      alpha[r] = al;
      // spill P (C layout) to per-wave scratch in (m,key) row-major
      const int m = r + half * 8;
      pw[m * KT + 0 * 16 + l16] = f2bf(p0);
      pw[m * KT + 1 * 16 + l16] = f2bf(p1);
    }

    // rescale accumulators by per-row alpha
#pragma unroll
    for (int t = 0; t < 8; ++t)
#pragma unroll
      for (int r = 0; r < 8; ++r) o[t][r] *= alpha[r];

    // reload P in A layout (16x32): two contiguous 8-elem runs per lane
    const __bf16* prow = pw + l16 * KT;
    const v16bf pa = load16bf(prow + half * 16, prow + half * 16 + 8);

    // ---- O += P @ V over this wave's 128 output columns (8 n-subtiles)
#pragma unroll
    for (int t = 0; t < 8; ++t) {
      // B[k][n] = V[key k][col n] = vT[col n][key k]: contiguous in vT
      const __bf16* vb =
          lds_vT + (wn * 128 + t * 16 + l16) * KT + half * 16;
      v16bf bm = load16bf(vb, vb + 8);
      o[t] = __builtin_amdgcn_wmma_f32_16x16x32_bf16(
          false, pa, false, bm, (short)0, o[t], false, false);
    }
  }

  // ---- normalize and store fp32 output
  const long orow0 = qrow0 + wm * 16;
#pragma unroll
  for (int r = 0; r < 8; ++r) {
    const float inv = 1.0f / l_s[r];
    const long row = orow0 + r + half * 8;
    float* op = out + row * D_DIM + wn * 128 + l16;
#pragma unroll
    for (int t = 0; t < 8; ++t) op[t * 16] = o[t][r] * inv;
  }
}

// ---------------- launcher ----------------
extern "C" void kernel_launch(void* const* d_in, const int* in_sizes, int n_in,
                              void* d_out, int out_size, void* d_ws, size_t ws_size,
                              hipStream_t stream) {
  const float* x  = (const float*)d_in[0];
  const float* Wq = (const float*)d_in[1];
  const float* bq = (const float*)d_in[2];
  const float* Wk = (const float*)d_in[3];
  const float* bk = (const float*)d_in[4];
  const float* Wv = (const float*)d_in[5];
  const float* bv = (const float*)d_in[6];
  float* out = (float*)d_out;

  // workspace layout (bf16): 3 weight mats + Q/K/V activations (~50.6 MB)
  char* ws = (char*)d_ws;
  const size_t wBytes = (size_t)D_DIM * D_DIM * sizeof(unsigned short);
  const size_t aBytes = (size_t)BATCH * S_LEN * D_DIM * sizeof(unsigned short);
  __bf16* wq_b = (__bf16*)ws;                 ws += wBytes;
  __bf16* wk_b = (__bf16*)ws;                 ws += wBytes;
  __bf16* wv_b = (__bf16*)ws;                 ws += wBytes;
  __bf16* q_b  = (__bf16*)ws;                 ws += aBytes;
  __bf16* k_b  = (__bf16*)ws;                 ws += aBytes;
  __bf16* v_b  = (__bf16*)ws;                 ws += aBytes;

  // 1) weights -> bf16
  cvt_w_kernel<<<(3 * D_DIM * D_DIM + 255) / 256, 256, 0, stream>>>(
      Wq, Wk, Wv, wq_b, wk_b, wv_b);

  // 2) fused QKV projection (WMMA)
  proj_kernel<<<BATCH * S_LEN / 16, 128, 0, stream>>>(
      x, wq_b, bq, wk_b, bk, wv_b, bv, q_b, k_b, v_b);

  // 3) flash attention (WMMA + online softmax)
  attn_kernel<<<dim3(BATCH, S_LEN / QT), 128, 0, stream>>>(
      q_b, k_b, v_b, out);
}
